// GlobalMamba_42975442763813
// MI455X (gfx1250) — matrix operand order repaired and checked by reading, not compiled, not run on hardware
//
#include <hip/hip_runtime.h>
#include <hip/hip_bf16.h>
#include <math.h>

typedef __attribute__((ext_vector_type(16))) __bf16 v16bf;
typedef __attribute__((ext_vector_type(8)))  float  v8f;

#define CB   4
#define CT   2048
#define CDIN 512
#define CDM  1024
#define CDO  1024
#define CL   2
#define CDS  16
#define CDC  4
#define CDI  2048   // EXP * DM
#define CDTR 64     // DM / 16
#define CM   (CB * CT)   // 8192 rows for every GEMM

#define CCH  128         // scan chunk length
#define CNCH (CT / CCH)  // 16 chunks
#define CBD  (CB * CDI)  // 8192 channels

#define TPAD 36          // padded row stride (floats) -> conflict-free ds reads

// ---------------------------------------------------------------------------
// CDNA5 async global->LDS copy (ASYNCcnt-tracked). 16B per lane.
// ---------------------------------------------------------------------------
__device__ __forceinline__ void async_b128(unsigned int lds_off, const float* gaddr) {
  asm volatile("global_load_async_to_lds_b128 %0, %1, off"
               :: "v"(lds_off), "v"(gaddr) : "memory");
}
__device__ __forceinline__ void wait_async6() {
  asm volatile("s_wait_asynccnt 6" ::: "memory");
}
__device__ __forceinline__ void wait_async0() {
  asm volatile("s_wait_asynccnt 0" ::: "memory");
}
__device__ __forceinline__ unsigned int lds_addr32(const void* p) {
  return (unsigned int)(unsigned long long)p;   // flat->LDS uses addr[31:0]
}

// ---------------------------------------------------------------------------
// GEMM: Y[m,n] = epi( sum_k X[m,k] * W[n,k] + bias[n] )
//   mode 0: (+bias).  mode 1: softplus(+bias).  mode 2: + R[m,n].
// Block: 8 waves, 128(M) x 64(N) tile. Wave: 32x32 (4 WMMA accumulators).
// A/B tiles staged f32 in LDS via async copies, double-buffered.
// ---------------------------------------------------------------------------
__global__ __launch_bounds__(256) void k_gemm_bf16(
    const float* __restrict__ X, int lda,
    const float* __restrict__ W,
    const float* __restrict__ bias,
    const float* __restrict__ R,
    float* __restrict__ Y, int ldy,
    int N, int K, int mode)
{
  __shared__ alignas(16) float sA[2][128 * TPAD];   // 36.0 KB
  __shared__ alignas(16) float sB[2][ 64 * TPAD];   // 18.0 KB

  const int tid   = threadIdx.x;
  const int lane  = tid & 31;
  const int wave  = tid >> 5;
  const int waveM = wave >> 1;              // 0..3 -> 32-row group
  const int waveN = wave & 1;               // 0..1 -> 32-col group
  const int rowBase0 = blockIdx.y * 128;
  const int colBase0 = blockIdx.x * 64;
  const int half = lane >> 4;
  const int r    = lane & 15;

  auto issue_stage = [&](int buf, int kb) {
#pragma unroll
    for (int i = 0; i < 4; ++i) {
      const int c   = tid + 256 * i;
      const int row = c >> 3;
      const int c4  = (c & 7) * 4;
      const float* g = X + (size_t)(rowBase0 + row) * (size_t)lda + kb + c4;
      async_b128(lds_addr32(&sA[buf][row * TPAD + c4]), g);
    }
#pragma unroll
    for (int i = 0; i < 2; ++i) {
      const int c   = tid + 256 * i;
      const int col = c >> 3;
      const int c4  = (c & 7) * 4;
      int gn = colBase0 + col;
      if (gn >= N) gn = N - 1;               // clamp: valid memory, zeroed later
      const float* g = W + (size_t)gn * (size_t)K + kb + c4;
      async_b128(lds_addr32(&sB[buf][col * TPAD + c4]), g);
    }
  };

  v8f acc[2][2];
#pragma unroll
  for (int m = 0; m < 2; ++m)
#pragma unroll
    for (int t = 0; t < 2; ++t)
#pragma unroll
      for (int i = 0; i < 8; ++i) acc[m][t][i] = 0.f;

  issue_stage(0, 0);

  int buf = 0;
  for (int kb = 0; kb < K; kb += 32) {
    const bool hasNext = (kb + 32) < K;
    if (hasNext) issue_stage(buf ^ 1, kb + 32);
    if (hasNext) wait_async6(); else wait_async0();
    __syncthreads();

    const float* As = sA[buf];
    const float* Bs = sB[buf];

    // A fragments: 16x32 bf16 ISA layout
    v16bf afrag[2];
#pragma unroll
    for (int m = 0; m < 2; ++m) {
      const int row = waveM * 32 + m * 16 + r;
      const float* p = As + row * TPAD + half * 8;
#pragma unroll
      for (int j = 0; j < 8; ++j) afrag[m][j]     = (__bf16)p[j];
#pragma unroll
      for (int j = 0; j < 8; ++j) afrag[m][8 + j] = (__bf16)p[16 + j];
    }

    // B fragments + WMMA
#pragma unroll
    for (int t = 0; t < 2; ++t) {
      const int coll = waveN * 32 + t * 16 + r;
      v16bf bm;
      if (colBase0 + coll < N) {
        const float* p = Bs + coll * TPAD + half * 8;
#pragma unroll
        for (int j = 0; j < 16; ++j) bm[j] = (__bf16)p[(j < 8) ? j : (8 + j)];
      } else {
#pragma unroll
        for (int j = 0; j < 16; ++j) bm[j] = (__bf16)0.f;
      }
      acc[0][t] = __builtin_amdgcn_wmma_f32_16x16x32_bf16(
          false, afrag[0], false, bm, (short)0, acc[0][t], false, false);
      acc[1][t] = __builtin_amdgcn_wmma_f32_16x16x32_bf16(
          false, afrag[1], false, bm, (short)0, acc[1][t], false, false);
    }

    __syncthreads();
    buf ^= 1;
  }

#pragma unroll
  for (int t = 0; t < 2; ++t) {
    const int n = colBase0 + waveN * 32 + t * 16 + r;
    if (n >= N) continue;
    const float bv = bias ? bias[n] : 0.f;
#pragma unroll
    for (int m = 0; m < 2; ++m) {
#pragma unroll
      for (int v = 0; v < 8; ++v) {
        const int mm = rowBase0 + waveM * 32 + m * 16 + half * 8 + v;
        float val = acc[m][t][v] + bv;
        if (mode == 1) {
          val = (val > 20.f) ? val : log1pf(expf(val));      // softplus
        } else if (mode == 2) {
          val += R[(size_t)mm * (size_t)ldy + n];             // residual add
        }
        Y[(size_t)mm * (size_t)ldy + n] = val;
      }
    }
  }
}

// ---------------------------------------------------------------------------
// LayerNorm over last dim D (one block per row)
// ---------------------------------------------------------------------------
__global__ __launch_bounds__(256) void k_layernorm(
    const float* __restrict__ x, const float* __restrict__ w,
    const float* __restrict__ b, float* __restrict__ y, int D)
{
  __shared__ float ssum[256];
  __shared__ float ssq[256];
  const int row = blockIdx.x;
  const float* xr = x + (size_t)row * D;
  float s = 0.f, q = 0.f;
  for (int i = threadIdx.x; i < D; i += 256) { float v = xr[i]; s += v; q += v * v; }
  ssum[threadIdx.x] = s; ssq[threadIdx.x] = q;
  __syncthreads();
  for (int off = 128; off > 0; off >>= 1) {
    if (threadIdx.x < off) {
      ssum[threadIdx.x] += ssum[threadIdx.x + off];
      ssq[threadIdx.x]  += ssq[threadIdx.x + off];
    }
    __syncthreads();
  }
  const float mu  = ssum[0] / (float)D;
  const float var = ssq[0] / (float)D - mu * mu;
  const float rst = rsqrtf(var + 1e-5f);
  float* yr = y + (size_t)row * D;
  for (int i = threadIdx.x; i < D; i += 256)
    yr[i] = (xr[i] - mu) * rst * w[i] + b[i];
}

// ---------------------------------------------------------------------------
// Depthwise causal conv (DC=4) + bias + SiLU.
// ---------------------------------------------------------------------------
__global__ __launch_bounds__(256) void k_conv_silu(
    const float* __restrict__ xz, const float* __restrict__ cw,
    const float* __restrict__ cb, float* __restrict__ u)
{
  const long long idx = (long long)blockIdx.x * 256 + threadIdx.x;
  if (idx >= (long long)CM * CDI) return;
  const int d = (int)(idx & (CDI - 1));
  const long long bt = idx / CDI;
  const int t = (int)(bt & (CT - 1));
  float acc = cb[d];
#pragma unroll
  for (int j = 0; j < CDC; ++j) {
    const int tt = t - (CDC - 1) + j;
    if (tt >= 0)
      acc += cw[d * CDC + j] * xz[(bt - (CDC - 1) + j) * (2 * CDI) + d];
  }
  const float sig = 1.f / (1.f + __expf(-acc));
  u[bt * CDI + d] = acc * sig;
}

// ---------------------------------------------------------------------------
// Chunked selective scan.
// Diagonal recurrence: h_t = exp(delta_t*A)*h_{t-1} + delta_t*u_t*B_t.
// Chunk decay = exp(A * sum(delta))  (product of per-step exps).
// Pass 1: per (chunk,b,d) -> chunk-local end state S (zero init) + sum(delta).
// Pass 2: per (b,d)       -> chain start states H across 16 chunks.
// Pass 3: per (chunk,b,d) -> exact recurrence from H + fused gate epilogue.
// ---------------------------------------------------------------------------
__global__ __launch_bounds__(256) void k_scan_part1(
    const float* __restrict__ u, const float* __restrict__ delta,
    const float* __restrict__ dbc, const float* __restrict__ A_log,
    float* __restrict__ S, float* __restrict__ Ssum)
{
  const int idx = blockIdx.x * 256 + threadIdx.x;   // (c, b, d), d fastest
  const int d = idx & (CDI - 1);
  const int b = (idx >> 11) & (CB - 1);
  const int c = idx >> 13;

  float A[CDS], h[CDS];
#pragma unroll
  for (int s = 0; s < CDS; ++s) { A[s] = -__expf(A_log[d * CDS + s]); h[s] = 0.f; }

  float sd = 0.f;
  const int t0 = c * CCH;
  for (int t = t0; t < t0 + CCH; ++t) {
    const size_t bt = (size_t)b * CT + t;
    const float dl = delta[bt * CDI + d];
    const float du = dl * u[bt * CDI + d];
    sd += dl;
    const float4* Bm = (const float4*)(dbc + bt * 96 + CDTR);
#pragma unroll
    for (int s4 = 0; s4 < 4; ++s4) {
      const float4 Bv = Bm[s4];
      float* hp = &h[s4 * 4];
      const float* Ap = &A[s4 * 4];
      hp[0] = hp[0] * __expf(dl * Ap[0]) + du * Bv.x;
      hp[1] = hp[1] * __expf(dl * Ap[1]) + du * Bv.y;
      hp[2] = hp[2] * __expf(dl * Ap[2]) + du * Bv.z;
      hp[3] = hp[3] * __expf(dl * Ap[3]) + du * Bv.w;
    }
  }
  float* Sp = S + (size_t)idx * CDS;
#pragma unroll
  for (int s = 0; s < CDS; ++s) Sp[s] = h[s];
  Ssum[idx] = sd;
}

__global__ __launch_bounds__(256) void k_scan_part2(
    const float* __restrict__ S, const float* __restrict__ Ssum,
    const float* __restrict__ A_log, float* __restrict__ H)
{
  const int j = blockIdx.x * 256 + threadIdx.x;     // (b, d)
  if (j >= CBD) return;
  const int d = j & (CDI - 1);

  float A[CDS], h[CDS];
#pragma unroll
  for (int s = 0; s < CDS; ++s) { A[s] = -__expf(A_log[d * CDS + s]); h[s] = 0.f; }

  for (int c = 0; c < CNCH; ++c) {
    const size_t base = ((size_t)c * CBD + j) * CDS;
    float* Hp = H + base;
#pragma unroll
    for (int s = 0; s < CDS; ++s) Hp[s] = h[s];      // start state of chunk c
    const float sd = Ssum[(size_t)c * CBD + j];
    const float* Sp = S + base;
#pragma unroll
    for (int s = 0; s < CDS; ++s)
      h[s] = h[s] * __expf(A[s] * sd) + Sp[s];
  }
}

__global__ __launch_bounds__(256) void k_scan_part3(
    const float* __restrict__ u, const float* __restrict__ delta,
    const float* __restrict__ dbc, const float* __restrict__ xz,
    const float* __restrict__ A_log, const float* __restrict__ Dp,
    const float* __restrict__ H, float* __restrict__ y)
{
  const int idx = blockIdx.x * 256 + threadIdx.x;   // (c, b, d)
  const int d = idx & (CDI - 1);
  const int b = (idx >> 11) & (CB - 1);
  const int c = idx >> 13;
  const int j = (b << 11) | d;

  float A[CDS], h[CDS];
#pragma unroll
  for (int s = 0; s < CDS; ++s) A[s] = -__expf(A_log[d * CDS + s]);
  const float* Hp = H + ((size_t)c * CBD + j) * CDS;
#pragma unroll
  for (int s = 0; s < CDS; ++s) h[s] = Hp[s];
  const float dskip = Dp[d];

  const int t0 = c * CCH;
  for (int t = t0; t < t0 + CCH; ++t) {
    const size_t bt = (size_t)b * CT + t;
    const float ut = u[bt * CDI + d];
    const float dl = delta[bt * CDI + d];
    const float4* Bm = (const float4*)(dbc + bt * 96 + CDTR);
    const float4* Cm = (const float4*)(dbc + bt * 96 + CDTR + CDS);
    const float du = dl * ut;
    float acc = 0.f;
#pragma unroll
    for (int s4 = 0; s4 < 4; ++s4) {
      const float4 Bv = Bm[s4];
      const float4 Cv = Cm[s4];
      float* hp = &h[s4 * 4];
      const float* Ap = &A[s4 * 4];
      hp[0] = hp[0] * __expf(dl * Ap[0]) + du * Bv.x; acc += hp[0] * Cv.x;
      hp[1] = hp[1] * __expf(dl * Ap[1]) + du * Bv.y; acc += hp[1] * Cv.y;
      hp[2] = hp[2] * __expf(dl * Ap[2]) + du * Bv.z; acc += hp[2] * Cv.z;
      hp[3] = hp[3] * __expf(dl * Ap[3]) + du * Bv.w; acc += hp[3] * Cv.w;
    }
    const float z = xz[bt * (2 * CDI) + CDI + d];
    const float sig = 1.f / (1.f + __expf(-z));
    y[bt * CDI + d] = (acc + ut * dskip) * (z * sig);
  }
}

// ---------------------------------------------------------------------------
// Masked mean over T. mask true => padding (excluded).
// ---------------------------------------------------------------------------
__global__ __launch_bounds__(256) void k_masked_mean(
    const float* __restrict__ xf, const unsigned char* __restrict__ mask,
    float* __restrict__ out)
{
  const int b = blockIdx.y;
  const int o = blockIdx.x * 256 + threadIdx.x;
  if (o >= CDO) return;
  float acc = 0.f, cnt = 0.f;
  for (int t = 0; t < CT; ++t) {
    if (!mask[b * CT + t]) {
      acc += xf[((size_t)b * CT + t) * CDO + o];
      cnt += 1.f;
    }
  }
  out[b * CDO + o] = acc / fmaxf(cnt, 1.f);
}

// ---------------------------------------------------------------------------
extern "C" void kernel_launch(void* const* d_in, const int* in_sizes, int n_in,
                              void* d_out, int out_size, void* d_ws, size_t ws_size,
                              hipStream_t stream)
{
  const float*         features  = (const float*)d_in[0];
  const unsigned char* mask      = (const unsigned char*)d_in[1];
  const float*         inp_w     = (const float*)d_in[2];
  const float*         inp_b     = (const float*)d_in[3];
  const float*         norm_w    = (const float*)d_in[4];
  const float*         norm_b    = (const float*)d_in[5];
  const float*         in_proj_w = (const float*)d_in[6];
  const float*         conv_w    = (const float*)d_in[7];
  const float*         conv_b    = (const float*)d_in[8];
  const float*         x_proj_w  = (const float*)d_in[9];
  const float*         dt_proj_w = (const float*)d_in[10];
  const float*         dt_proj_b = (const float*)d_in[11];
  const float*         A_log     = (const float*)d_in[12];
  const float*         Dskip     = (const float*)d_in[13];
  const float*         mout_w    = (const float*)d_in[14];
  const float*         out_w     = (const float*)d_in[15];
  const float*         out_b     = (const float*)d_in[16];
  float*               out       = (float*)d_out;

  // workspace carve-up (floats)
  float* ws = (float*)d_ws;
  float* buf_x     = ws;                                   // (M, DM)
  float* buf_xn    = buf_x    + (size_t)CM * CDM;          // (M, DM)
  float* buf_xz    = buf_xn   + (size_t)CM * CDM;          // (M, 2*DI)
  float* buf_u     = buf_xz   + (size_t)CM * 2 * CDI;      // (M, DI)
  float* buf_dbc   = buf_u    + (size_t)CM * CDI;          // (M, 96)
  float* buf_delta = buf_dbc  + (size_t)CM * 96;           // (M, DI)
  float* buf_y     = buf_delta+ (size_t)CM * CDI;          // (M, DI)
  float* buf_fin   = buf_y    + (size_t)CM * CDI;          // (M, DO)
  float* buf_S     = buf_fin  + (size_t)CM * CDO;          // (NCH, B*DI, DS)
  float* buf_H     = buf_S    + (size_t)CNCH * CBD * CDS;  // (NCH, B*DI, DS)
  float* buf_Ssum  = buf_H    + (size_t)CNCH * CBD * CDS;  // (NCH, B*DI)

  const dim3 blk(256);
  auto ggrid = [](int N) { return dim3((N + 63) / 64, CM / 128); };
  const dim3 scgrid((CNCH * CBD) / 256);

  // x = features @ inp_w.T + inp_b
  k_gemm_bf16<<<ggrid(CDM), blk, 0, stream>>>(
      features, CDIN, inp_w, inp_b, nullptr, buf_x, CDM, CDM, CDIN, 0);

  for (int l = 0; l < CL; ++l) {
    k_layernorm<<<dim3(CM), blk, 0, stream>>>(
        buf_x, norm_w + (size_t)l * CDM, norm_b + (size_t)l * CDM, buf_xn, CDM);

    // xz = xn @ in_proj_w[l].T   (N = 2*DI)
    k_gemm_bf16<<<ggrid(2 * CDI), blk, 0, stream>>>(
        buf_xn, CDM, in_proj_w + (size_t)l * 2 * CDI * CDM, nullptr, nullptr,
        buf_xz, 2 * CDI, 2 * CDI, CDM, 0);

    // u = silu(conv(xi) + cb)
    k_conv_silu<<<dim3(((size_t)CM * CDI + 255) / 256), blk, 0, stream>>>(
        buf_xz, conv_w + (size_t)l * CDI * CDC, conv_b + (size_t)l * CDI, buf_u);

    // dbc = u @ x_proj_w[l].T    (N = 96)
    k_gemm_bf16<<<ggrid(CDTR + 2 * CDS), blk, 0, stream>>>(
        buf_u, CDI, x_proj_w + (size_t)l * (CDTR + 2 * CDS) * CDI, nullptr, nullptr,
        buf_dbc, CDTR + 2 * CDS, CDTR + 2 * CDS, CDI, 0);

    // delta = softplus(dt @ dt_proj_w[l].T + dt_proj_b[l])  (K = 64, lda = 96)
    k_gemm_bf16<<<ggrid(CDI), blk, 0, stream>>>(
        buf_dbc, CDTR + 2 * CDS, dt_proj_w + (size_t)l * CDI * CDTR,
        dt_proj_b + (size_t)l * CDI, nullptr, buf_delta, CDI, CDI, CDTR, 1);

    // chunked selective scan + gate
    k_scan_part1<<<scgrid, blk, 0, stream>>>(
        buf_u, buf_delta, buf_dbc, A_log + (size_t)l * CDI * CDS, buf_S, buf_Ssum);
    k_scan_part2<<<dim3(CBD / 256), blk, 0, stream>>>(
        buf_S, buf_Ssum, A_log + (size_t)l * CDI * CDS, buf_H);
    k_scan_part3<<<scgrid, blk, 0, stream>>>(
        buf_u, buf_delta, buf_dbc, buf_xz, A_log + (size_t)l * CDI * CDS,
        Dskip + (size_t)l * CDI, buf_H, buf_y);

    // x = y @ mout_w[l].T + residual(x)
    k_gemm_bf16<<<ggrid(CDM), blk, 0, stream>>>(
        buf_y, CDI, mout_w + (size_t)l * CDM * CDI, nullptr, buf_x,
        buf_x, CDM, CDM, CDI, 2);
  }

  // final projection
  k_gemm_bf16<<<ggrid(CDO), blk, 0, stream>>>(
      buf_x, CDM, out_w, out_b, nullptr, buf_fin, CDO, CDO, CDM, 0);

  // masked mean over T
  k_masked_mean<<<dim3((CDO + 255) / 256, CB), blk, 0, stream>>>(
      buf_fin, mask, out);
}
